// ReportDecoder_13185549599223
// MI455X (gfx1250) — matrix operand order repaired and checked by reading, not compiled, not run on hardware
//
#include <hip/hip_runtime.h>
#include <hip/hip_bf16.h>
#include <math.h>

// ---------------------------------------------------------------------------
// Problem constants (from reference)
// ---------------------------------------------------------------------------
namespace cfg {
constexpr int B  = 256;          // batch
constexpr int S  = 256;          // image sequence
constexpr int E  = 1024;         // embedding
constexpr int H  = 1024;         // hidden
constexpr int VP = 8;
constexpr int L  = 64;
constexpr int T  = 16;           // decode steps
constexpr int IN   = E + VP + 1 + L;   // 1097 lstm input width
constexpr int KPIN = 1152;             // IN padded to multiple of 64 (pipelined K loop)
constexpr int NG   = 4 * H;            // 4096 gate width
constexpr int NFC  = L + H + 1 + 1;    // 1090 fc out
constexpr int NFCP = 1152;             // fc out padded to multiple of 64
constexpr int OUTC = H + 2;            // 1026 per-step output width
}

// ---------------------------------------------------------------------------
// WMMA types (gfx1250, wave32)
// ---------------------------------------------------------------------------
typedef __attribute__((ext_vector_type(16))) __bf16       v16bf;
typedef __attribute__((ext_vector_type(8)))  float        v8f;
typedef __attribute__((ext_vector_type(4)))  unsigned int v4u;

union Frag16 { v16bf v; v4u q[2]; };   // 32 bytes: one bf16 A/B fragment per lane

__device__ __forceinline__ unsigned short f32_to_bf16u(float f) {
  unsigned int u = __float_as_uint(f);
  u += 0x7FFFu + ((u >> 16) & 1u);     // round-to-nearest-even
  return (unsigned short)(u >> 16);
}
__device__ __forceinline__ float sigmoidf_(float x) {
  return 1.0f / (1.0f + __expf(-x));
}

// ---------------------------------------------------------------------------
// image mean over S:  (B,S,E) f32 -> (B,E) f32 + bf16 copy.  Pure HBM stream.
// ---------------------------------------------------------------------------
__global__ __launch_bounds__(256) void mean_kernel(
    const float* __restrict__ image, float* __restrict__ mean_f,
    unsigned short* __restrict__ mean_h) {
  int idx = blockIdx.x * blockDim.x + threadIdx.x;       // over B*E
  int b = idx >> 10, e = idx & (cfg::E - 1);
  const float* p = image + (size_t)b * cfg::S * cfg::E + e;
  float s = 0.f;
  #pragma unroll 4
  for (int t = 0; t < cfg::S; ++t) s += p[(size_t)t * cfg::E];
  float m = s * (1.0f / cfg::S);
  mean_f[idx] = m;
  mean_h[idx] = f32_to_bf16u(m);
}

// ---------------------------------------------------------------------------
// f32 (srows x scols) -> bf16 (drows x dcols), zero padded
// ---------------------------------------------------------------------------
__global__ __launch_bounds__(256) void conv_pad_bf16(
    const float* __restrict__ src, unsigned short* __restrict__ dst,
    int srows, int scols, int drows, int dcols) {
  size_t idx = (size_t)blockIdx.x * blockDim.x + threadIdx.x;
  size_t total = (size_t)drows * dcols;
  if (idx >= total) return;
  int r = (int)(idx / dcols), c = (int)(idx % dcols);
  float v = (r < srows && c < scols) ? src[(size_t)r * scols + c] : 0.f;
  dst[idx] = f32_to_bf16u(v);
}

// ---------------------------------------------------------------------------
// Build X (T*B, KPIN) bf16:  row t*B+b = [mean_b | vp_b | begin_t | label_{b,t} | 0pad]
// ---------------------------------------------------------------------------
__global__ __launch_bounds__(256) void pack_x(
    const float* __restrict__ mean_f, const float* __restrict__ vp,
    const float* __restrict__ label, unsigned short* __restrict__ X) {
  size_t idx = (size_t)blockIdx.x * blockDim.x + threadIdx.x;
  size_t total = (size_t)cfg::T * cfg::B * cfg::KPIN;
  if (idx >= total) return;
  int c = (int)(idx % cfg::KPIN);
  int r = (int)(idx / cfg::KPIN);
  int t = r / cfg::B, b = r % cfg::B;
  float v;
  if (c < cfg::E)                v = mean_f[(size_t)b * cfg::E + c];
  else if (c < cfg::E + cfg::VP) v = vp[(size_t)b * cfg::VP + (c - cfg::E)];
  else if (c == cfg::E + cfg::VP) v = (t == 0) ? 1.f : 0.f;
  else if (c < cfg::IN)          v = label[((size_t)b * (cfg::T + 1) + t) * cfg::L
                                           + (c - (cfg::E + cfg::VP + 1))];
  else                           v = 0.f;
  X[idx] = f32_to_bf16u(v);
}

// bias helpers
__global__ void vec_add2(const float* a, const float* b, float* o, int n) {
  int i = blockIdx.x * blockDim.x + threadIdx.x;
  if (i < n) o[i] = a[i] + b[i];
}
__global__ void vec_pad(const float* a, float* o, int n, int npad) {
  int i = blockIdx.x * blockDim.x + threadIdx.x;
  if (i < npad) o[i] = (i < n) ? a[i] : 0.f;
}

// ---------------------------------------------------------------------------
// Fragment load / MMA helpers (ISA 7.12.2 VGPR layouts, wave32)
//   A 16x32 frag, lane=(half,row lm): e=0..7 -> k=8*half+e ; e=8..15 -> 16+8*half+(e-8)
//   B 32x16 frag, lane=(half,col lm): same k order as A ;  B[k][n] = W[n][k]
// ---------------------------------------------------------------------------
__device__ __forceinline__ void load_a2(Frag16 (&a)[2],
    const unsigned short* __restrict__ arow0, const unsigned short* __restrict__ arow1,
    int k, int half) {
  a[0].q[0] = *(const v4u*)(arow0 + k + 8 * half);
  a[0].q[1] = *(const v4u*)(arow0 + k + 16 + 8 * half);
  a[1].q[0] = *(const v4u*)(arow1 + k + 8 * half);
  a[1].q[1] = *(const v4u*)(arow1 + k + 16 + 8 * half);
}

__device__ __forceinline__ void load_b4(Frag16 (&b)[4],
    const unsigned short* __restrict__ wbase, int ldw, int k, int half) {
  #pragma unroll
  for (int s = 0; s < 4; ++s) {
    const unsigned short* wr = wbase + (size_t)(s * 16) * ldw + k + 8 * half;
    b[s].q[0] = *(const v4u*)(wr);
    b[s].q[1] = *(const v4u*)(wr + 16);
  }
}

__device__ __forceinline__ void mma8(const Frag16 (&a)[2], const Frag16 (&b)[4],
                                     v8f (&acc)[2][4]) {
  #pragma unroll
  for (int r = 0; r < 2; ++r)
    #pragma unroll
    for (int s = 0; s < 4; ++s)
      acc[r][s] = __builtin_amdgcn_wmma_f32_16x16x32_bf16(
          false, a[r].v, false, b[s].v, (short)0, acc[r][s], false, false);
}

// ---------------------------------------------------------------------------
// WMMA bf16 GEMM:  Cout(MxN) = act( A(MxK) * W(NxK)^T + Cin + bias )
//   * one wave computes a 32(M) x 64(N) tile: 8 WMMAs per 32-wide K step
//   * software-pipelined: ping-pong fragment registers, loads overlap WMMA
//   * 8 waves / block share the same 32 A rows (WGP$ reuse)
//   * M % 32 == 0, K % 64 == 0 (K >= 128), N padded to 64
// ---------------------------------------------------------------------------
__global__ __launch_bounds__(256) void wmma_gemm_bf16(
    const unsigned short* __restrict__ A, int lda,
    const unsigned short* __restrict__ W, int ldw,
    const float* __restrict__ Cin, int ldcin,       // optional additive base
    const float* __restrict__ bias,                 // optional, length >= Npad
    float* __restrict__ Cout, int ldc,              // optional f32 out
    unsigned short* __restrict__ Hout, int ldh,     // optional bf16 out
    int M, int Npad, int K, int act)                // act: 0 none, 1 tanh
{
  const int lane = threadIdx.x & 31;
  const int wave = threadIdx.x >> 5;
  const int m0 = blockIdx.y * 32;
  const int n_strip = (blockIdx.x * 8 + wave) * 64;
  if (n_strip >= Npad || m0 >= M) return;

  const int half = lane >> 4;      // 0: lanes 0-15, 1: lanes 16-31
  const int lm   = lane & 15;      // A row in sub-tile; also N column for B/C/D

  v8f acc[2][4];
  #pragma unroll
  for (int r = 0; r < 2; ++r) {
    #pragma unroll
    for (int s = 0; s < 4; ++s) {
      if (Cin) {
        const float* cp = Cin + (size_t)(m0 + 16 * r) * ldcin + n_strip + s * 16 + lm;
        #pragma unroll
        for (int v = 0; v < 8; ++v) acc[r][s][v] = cp[(size_t)(v + 8 * half) * ldcin];
      } else {
        #pragma unroll
        for (int v = 0; v < 8; ++v) acc[r][s][v] = 0.f;
      }
    }
  }

  const unsigned short* arow0 = A + (size_t)(m0 + lm) * lda;
  const unsigned short* arow1 = arow0 + (size_t)16 * lda;
  const unsigned short* wbase = W + (size_t)(n_strip + lm) * ldw;

  Frag16 a0[2], a1[2], b0[4], b1[4];
  load_a2(a0, arow0, arow1, 0, half);
  load_b4(b0, wbase, ldw, 0, half);

  int k0 = 0;
  for (; k0 + 64 < K; k0 += 64) {
    load_a2(a1, arow0, arow1, k0 + 32, half);   // prefetch odd stage
    load_b4(b1, wbase, ldw, k0 + 32, half);
    mma8(a0, b0, acc);                          // consume even stage
    load_a2(a0, arow0, arow1, k0 + 64, half);   // prefetch next even stage
    load_b4(b0, wbase, ldw, k0 + 64, half);
    mma8(a1, b1, acc);                          // consume odd stage
  }
  // tail: k0 == K - 64
  load_a2(a1, arow0, arow1, k0 + 32, half);
  load_b4(b1, wbase, ldw, k0 + 32, half);
  mma8(a0, b0, acc);
  mma8(a1, b1, acc);

  // C/D layout: VGPR v holds row m = 16*r + v + 8*half, column n = lm
  #pragma unroll
  for (int r = 0; r < 2; ++r) {
    #pragma unroll
    for (int s = 0; s < 4; ++s) {
      const int n = n_strip + s * 16 + lm;
      const float bv = bias ? bias[n] : 0.f;
      #pragma unroll
      for (int v = 0; v < 8; ++v) {
        const int m = m0 + 16 * r + v + 8 * half;
        float res = acc[r][s][v] + bv;
        if (act == 1) res = tanhf(res);
        if (Cout) Cout[(size_t)m * ldc + n] = res;
        if (Hout) Hout[(size_t)m * ldh + n] = f32_to_bf16u(res);
      }
    }
  }
}

// ---------------------------------------------------------------------------
// LSTM elementwise: gates (B,4H) f32, m (B,H) f32 in/out, h2 -> bf16 (B,H)
// ---------------------------------------------------------------------------
__global__ __launch_bounds__(256) void lstm_elem(
    const float* __restrict__ gates, float* __restrict__ m,
    unsigned short* __restrict__ h_bf16) {
  int idx = blockIdx.x * blockDim.x + threadIdx.x;   // over B*H
  int b = idx >> 10, hh = idx & (cfg::H - 1);
  const float* g = gates + (size_t)b * cfg::NG;
  float gi = sigmoidf_(g[hh]);
  float gf = sigmoidf_(g[cfg::H + hh]);
  float gg = tanhf(g[2 * cfg::H + hh]);
  float go = sigmoidf_(g[3 * cfg::H + hh]);
  float m2 = gf * m[idx] + gi * gg;
  m[idx] = m2;
  h_bf16[idx] = f32_to_bf16u(go * tanhf(m2));
}

// ---------------------------------------------------------------------------
// Head epilogue: fc (B, NFCP) f32 -> out[t] (B, 1026) = mask*[relu|sigmoid|exp]
// ---------------------------------------------------------------------------
__global__ __launch_bounds__(256) void epilogue(
    const float* __restrict__ fc, const int* __restrict__ length,
    float* __restrict__ out, int t) {
  int idx = blockIdx.x * blockDim.x + threadIdx.x;   // over B*OUTC
  if (idx >= cfg::B * cfg::OUTC) return;
  int b = idx / cfg::OUTC, j = idx % cfg::OUTC;
  const float* row = fc + (size_t)b * cfg::NFCP;
  float v;
  if (j < cfg::H)        { v = row[cfg::L + j]; v = v > 0.f ? v : 0.f; }
  else if (j == cfg::H)  { v = sigmoidf_(row[cfg::L + cfg::H]); }
  else                   { v = __expf(row[cfg::L + cfg::H + 1]); }
  float mask = (t < length[b] - 1) ? 1.f : 0.f;
  out[(size_t)t * cfg::B * cfg::OUTC + idx] = v * mask;
}

// ---------------------------------------------------------------------------
// Launcher
// ---------------------------------------------------------------------------
extern "C" void kernel_launch(void* const* d_in, const int* in_sizes, int n_in,
                              void* d_out, int out_size, void* d_ws, size_t ws_size,
                              hipStream_t stream) {
  using namespace cfg;
  const float* image  = (const float*)d_in[0];
  const float* vp     = (const float*)d_in[1];
  const float* label  = (const float*)d_in[2];
  const int*   length = (const int*)d_in[3];
  const float* W_h  = (const float*)d_in[4];
  const float* b_h  = (const float*)d_in[5];
  const float* W_m  = (const float*)d_in[6];
  const float* b_m  = (const float*)d_in[7];
  const float* W_ih = (const float*)d_in[8];
  const float* W_hh = (const float*)d_in[9];
  const float* b_ih = (const float*)d_in[10];
  const float* b_hh = (const float*)d_in[11];
  const float* W_fc = (const float*)d_in[12];
  const float* b_fc = (const float*)d_in[13];
  float* out = (float*)d_out;

  // ---- carve workspace (256B aligned) ----
  char* p = (char*)d_ws;
  auto carve = [&](size_t bytes) -> void* {
    void* r = (void*)p; p += (bytes + 255) & ~(size_t)255; return r;
  };
  float*          meanF = (float*)         carve((size_t)B * E * 4);
  unsigned short* meanH = (unsigned short*)carve((size_t)B * E * 2);
  unsigned short* WhH   = (unsigned short*)carve((size_t)H * E * 2);
  unsigned short* WmH   = (unsigned short*)carve((size_t)H * E * 2);
  unsigned short* WihH  = (unsigned short*)carve((size_t)NG * KPIN * 2);
  unsigned short* WhhH  = (unsigned short*)carve((size_t)NG * H * 2);
  unsigned short* WfcH  = (unsigned short*)carve((size_t)NFCP * H * 2);
  unsigned short* Xh    = (unsigned short*)carve((size_t)T * B * KPIN * 2);
  float*          Gin   = (float*)         carve((size_t)T * B * NG * 4);
  float*          gates = (float*)         carve((size_t)B * NG * 4);
  unsigned short* hH    = (unsigned short*)carve((size_t)B * H * 2);
  float*          mF    = (float*)         carve((size_t)B * H * 4);
  float*          fcF   = (float*)         carve((size_t)B * NFCP * 4);
  float*          biasG = (float*)         carve((size_t)NG * 4);
  float*          bfcP  = (float*)         carve((size_t)NFCP * 4);

  const int TB = 256;
  auto blks = [](size_t n, int tb) { return (unsigned)((n + tb - 1) / tb); };

  // ---- stage 0: mean + weight conversion (all parallel-independent) ----
  mean_kernel<<<blks((size_t)B * E, TB), TB, 0, stream>>>(image, meanF, meanH);
  conv_pad_bf16<<<blks((size_t)H * E, TB), TB, 0, stream>>>(W_h, WhH, H, E, H, E);
  conv_pad_bf16<<<blks((size_t)H * E, TB), TB, 0, stream>>>(W_m, WmH, H, E, H, E);
  conv_pad_bf16<<<blks((size_t)NG * KPIN, TB), TB, 0, stream>>>(W_ih, WihH, NG, IN, NG, KPIN);
  conv_pad_bf16<<<blks((size_t)NG * H, TB), TB, 0, stream>>>(W_hh, WhhH, NG, H, NG, H);
  conv_pad_bf16<<<blks((size_t)NFCP * H, TB), TB, 0, stream>>>(W_fc, WfcH, NFC, H, NFCP, H);
  vec_add2<<<blks(NG, TB), TB, 0, stream>>>(b_ih, b_hh, biasG, NG);
  vec_pad<<<blks(NFCP, TB), TB, 0, stream>>>(b_fc, bfcP, NFC, NFCP);
  pack_x<<<blks((size_t)T * B * KPIN, TB), TB, 0, stream>>>(meanF, vp, label, Xh);

  // ---- stage 1: h0 = tanh(mean @ W_h^T + b_h) -> bf16 ; m0 -> f32 ----
  {
    dim3 g((H + 511) / 512, B / 32);
    wmma_gemm_bf16<<<g, 256, 0, stream>>>(meanH, E, WhH, E, nullptr, 0, b_h,
                                          nullptr, 0, hH, H, B, H, E, 1);
    wmma_gemm_bf16<<<g, 256, 0, stream>>>(meanH, E, WmH, E, nullptr, 0, b_m,
                                          mF, H, nullptr, 0, B, H, E, 1);
  }

  // ---- stage 2: precompute input gates for ALL steps in one parallel GEMM
  //      Gin(T*B, 4H) = X @ W_ih^T + (b_ih + b_hh) ----
  {
    dim3 g((NG + 511) / 512, (T * B) / 32);
    wmma_gemm_bf16<<<g, 256, 0, stream>>>(Xh, KPIN, WihH, KPIN, nullptr, 0, biasG,
                                          Gin, NG, nullptr, 0, T * B, NG, KPIN, 0);
  }

  // ---- stage 3: sequential recurrence (only h @ W_hh^T on critical path) ----
  dim3 g3((NG + 511) / 512, B / 32);
  dim3 g4((NFCP + 511) / 512, B / 32);
  for (int t = 0; t < T; ++t) {
    // gates = Gin[t] + h @ W_hh^T
    wmma_gemm_bf16<<<g3, 256, 0, stream>>>(hH, H, WhhH, H,
                                           Gin + (size_t)t * B * NG, NG, nullptr,
                                           gates, NG, nullptr, 0, B, NG, H, 0);
    lstm_elem<<<blks((size_t)B * H, TB), TB, 0, stream>>>(gates, mF, hH);
    // fc = h2 @ W_fc^T + b_fc
    wmma_gemm_bf16<<<g4, 256, 0, stream>>>(hH, H, WfcH, H, nullptr, 0, bfcP,
                                           fcF, NFCP, nullptr, 0, B, NFCP, H, 0);
    epilogue<<<blks((size_t)B * OUTC, TB), TB, 0, stream>>>(fcF, length, out, t);
  }
}
